// SelfAttention_11699490914916
// MI455X (gfx1250) — compile-verified, matrix operand order checked by reading, not run on hardware
//
#include <hip/hip_runtime.h>
#include <stdint.h>

typedef __bf16 bf16;
typedef bf16  v16bf __attribute__((ext_vector_type(16)));
typedef float v8f   __attribute__((ext_vector_type(8)));

#define TM 128
#define TN 128
#define TK 32
#define LDST 40          // LDS row stride in bf16 elements
#define LDSB (LDST * 2)  // 80 bytes per row (16B aligned, bank-spread)

// ---------- bf16 helpers (RNE) ----------
__device__ inline uint16_t f2bfu(float f) {
    union { float f; uint32_t u; } a; a.f = f;
    uint32_t r = a.u + 0x7FFFu + ((a.u >> 16) & 1u);
    return (uint16_t)(r >> 16);
}
__device__ inline bf16 f2bf(float f) {
    union { uint16_t s; bf16 b; } o; o.s = f2bfu(f); return o.b;
}
__device__ inline uint32_t packbf(float lo, float hi) {
    return (uint32_t)f2bfu(lo) | ((uint32_t)f2bfu(hi) << 16);
}

// ---------- CDNA5 async global->LDS (ASYNCcnt-tracked DMA, no VGPR staging) ----------
__device__ inline void async_b128(uint32_t lds_byte_addr, const void* g) {
    asm volatile("global_load_async_to_lds_b128 %0, %1, off"
                 :: "v"(lds_byte_addr), "v"(g) : "memory");
}
__device__ inline void wait_async_le4() { asm volatile("s_wait_asynccnt 0x4" ::: "memory"); }
__device__ inline void wait_async_0()   { asm volatile("s_wait_asynccnt 0x0" ::: "memory"); }

// ---------- WMMA fragment load (ISA 7.12.2: 16-bit A 16x32 layout) ----------
__device__ inline v16bf frag_ld(const bf16* __restrict__ base, int lane) {
    const int r  = lane & 15;
    const int kh = (lane >> 4) & 1;
    const bf16* p = base + r * LDST + kh * 8;
    v16bf v;
#pragma unroll
    for (int i = 0; i < 8; ++i) v[i] = p[i];
#pragma unroll
    for (int i = 0; i < 8; ++i) v[8 + i] = p[16 + i];
    return v;
}

__device__ inline v8f wmma_bf16(v16bf a, v16bf b, v8f c) {
    return __builtin_amdgcn_wmma_f32_16x16x32_bf16(
        false, a, false, b, (short)0, c, false, false);
}

enum { EPI_BIAS_BF16 = 0, EPI_SCORES = 1, EPI_BF16 = 2, EPI_BIAS_F32 = 3, EPI_BIAS_BF16_T = 4 };

// ---------------- double-buffered async WMMA GEMM: C = A(MxK) * B^T(NxK) ----------------
// A: (M,K) row-major bf16.  B: (N,K) row-major bf16 (always N-major).
template <int EPI, bool CAUSAL>
__global__ __launch_bounds__(256)
void gemm_bf16(const bf16* __restrict__ Ag, const bf16* __restrict__ Bg,
               const float* __restrict__ bias,
               float* __restrict__ outF, bf16* __restrict__ outB,
               int K, int lda, int ldb, int ldc,
               long strideA, long strideB, long strideC, float scale)
{
    __shared__ __align__(16) bf16 sA[2][TM * LDST];
    __shared__ __align__(16) bf16 sB[2][TN * LDST];

    const int z = blockIdx.z;
    Ag += (long)z * strideA;
    Bg += (long)z * strideB;
    if (outF) outF += (long)z * strideC;
    if (outB) outB += (long)z * strideC;

    const int mBase = blockIdx.y * TM;
    const int nBase = blockIdx.x * TN;
    const int tid  = threadIdx.x;
    const int lane = tid & 31;
    const int wid  = tid >> 5;          // 8 waves
    const int wm = wid >> 2;            // 0..1 (64 rows)
    const int wn = wid & 3;             // 0..3 (32 cols)

    const bool fullMask = CAUSAL && (nBase > mBase + (TM - 1));

    v8f acc[4][2];
#pragma unroll
    for (int mt = 0; mt < 4; ++mt)
#pragma unroll
        for (int nt = 0; nt < 2; ++nt)
#pragma unroll
            for (int r = 0; r < 8; ++r) acc[mt][nt][r] = 0.0f;

    if (!fullMask) {
        // per-wave async tile slice: wave owns rows [wid*16, wid*16+16) of both tiles;
        // 2 instructions per tile (8 rows each), lane -> row = lane>>2, 16B chunk = lane&3.
        const int rsub  = (lane >> 2);        // 0..7
        const int csub  = (lane & 3) * 16;    // byte offset within 64B row
        const long ldaB = (long)lda * 2;
        const long ldbB = (long)ldb * 2;
        const char* gA0 = (const char*)(Ag + (long)(mBase + wid * 16 + rsub) * lda) + csub;
        const char* gB0 = (const char*)(Bg + (long)(nBase + wid * 16 + rsub) * ldb) + csub;
        const uint32_t lA0 = (uint32_t)(uintptr_t)&sA[0][(wid * 16 + rsub) * LDST] + csub;
        const uint32_t lB0 = (uint32_t)(uintptr_t)&sB[0][(wid * 16 + rsub) * LDST] + csub;
        const uint32_t bufOff = (uint32_t)(TM * LDSB);   // bytes between buffers

        #define ISSUE(bufi, k0)                                                   \
            do {                                                                  \
                const long kb = (long)(k0) * 2;                                   \
                const uint32_t lo = (bufi) * bufOff;                              \
                async_b128(lA0 + lo,            gA0 + kb);                        \
                async_b128(lA0 + lo + 8 * LDSB, gA0 + kb + 8 * ldaB);             \
                async_b128(lB0 + lo,            gB0 + kb);                        \
                async_b128(lB0 + lo + 8 * LDSB, gB0 + kb + 8 * ldbB);             \
            } while (0)

        const int NT = K / TK;
        ISSUE(0, 0);
        for (int kt = 0; kt < NT; ++kt) {
            const int cur = kt & 1;
            if (kt + 1 < NT) {
                ISSUE(cur ^ 1, (kt + 1) * TK);
                wait_async_le4();            // current buffer's 4 loads complete
            } else {
                wait_async_0();
            }
            __syncthreads();                 // all waves' DMA into cur visible

            const bf16* aw = &sA[cur][(wm * 64) * LDST];
            const bf16* bw = &sB[cur][(wn * 32) * LDST];
            v16bf af[4], bfv[2];
#pragma unroll
            for (int mt = 0; mt < 4; ++mt) af[mt]  = frag_ld(aw + mt * 16 * LDST, lane);
#pragma unroll
            for (int nt = 0; nt < 2; ++nt) bfv[nt] = frag_ld(bw + nt * 16 * LDST, lane);
#pragma unroll
            for (int mt = 0; mt < 4; ++mt)
#pragma unroll
                for (int nt = 0; nt < 2; ++nt)
                    acc[mt][nt] = wmma_bf16(af[mt], bfv[nt], acc[mt][nt]);
            __syncthreads();                 // cur fully consumed -> may be overwritten
        }
        #undef ISSUE
    }

    // ---- Epilogue (C/D layout: m = r + 8*(lane>>4), n = lane&15) ----
    const int ln = lane & 15;
    const int lh = lane >> 4;
#pragma unroll
    for (int mt = 0; mt < 4; ++mt)
#pragma unroll
        for (int nt = 0; nt < 2; ++nt)
#pragma unroll
            for (int r = 0; r < 8; ++r) {
                const int m = mBase + wm * 64 + mt * 16 + r + 8 * lh;
                const int n = nBase + wn * 32 + nt * 16 + ln;
                float v = fullMask ? 0.0f : acc[mt][nt][r];
                if (EPI == EPI_SCORES) {
                    v *= scale;
                    if (CAUSAL && (fullMask || n > m)) v = -1e9f;
                    outF[(long)m * ldc + n] = v;
                } else if (EPI == EPI_BIAS_BF16) {
                    outB[(long)m * ldc + n] = f2bf(v + bias[n]);
                } else if (EPI == EPI_BIAS_BF16_T) {      // write C^T (for V^T)
                    outB[(long)n * ldc + m] = f2bf(v + bias[n]);
                } else if (EPI == EPI_BF16) {
                    outB[(long)m * ldc + n] = f2bf(v);
                } else {                                  // EPI_BIAS_F32
                    outF[(long)m * ldc + n] = v + bias[n];
                }
            }
}

// ---------------- fp32 -> bf16 bulk convert (float4 -> uint2) ----------------
__global__ __launch_bounds__(256)
void cvt4(const float* __restrict__ s, bf16* __restrict__ d, long n4)
{
    const long i = (long)blockIdx.x * 256 + threadIdx.x;
    if (i < n4) {
        const float4 v = reinterpret_cast<const float4*>(s)[i];
        uint2 u; u.x = packbf(v.x, v.y); u.y = packbf(v.z, v.w);
        reinterpret_cast<uint2*>(d)[i] = u;
    }
}

// ---------------- row softmax: fp32 in-place + bf16 copy ----------------
__global__ __launch_bounds__(256)
void softmax_rows(float* __restrict__ attnF, bf16* __restrict__ attnB, int S)
{
    __shared__ float red[256];
    const long row = blockIdx.x;
    float* p  = attnF + row * (long)S;
    bf16*  pb = attnB + row * (long)S;
    const int tid = threadIdx.x;

    float x[8];
    float mx = -3.0e38f;
#pragma unroll
    for (int j = 0; j < 8; ++j) {
        x[j] = p[tid + j * 256];
        mx = fmaxf(mx, x[j]);
    }
    red[tid] = mx; __syncthreads();
#pragma unroll
    for (int s = 128; s > 0; s >>= 1) {
        if (tid < s) red[tid] = fmaxf(red[tid], red[tid + s]);
        __syncthreads();
    }
    mx = red[0]; __syncthreads();

    float sum = 0.0f;
#pragma unroll
    for (int j = 0; j < 8; ++j) {
        x[j] = __expf(x[j] - mx);      // masked (-1e9) entries underflow to 0
        sum += x[j];
    }
    red[tid] = sum; __syncthreads();
#pragma unroll
    for (int s = 128; s > 0; s >>= 1) {
        if (tid < s) red[tid] += red[tid + s];
        __syncthreads();
    }
    const float inv = 1.0f / red[0];
#pragma unroll
    for (int j = 0; j < 8; ++j) {
        const float v = x[j] * inv;
        p[tid + j * 256]  = v;
        pb[tid + j * 256] = f2bf(v);
    }
}

extern "C" void kernel_launch(void* const* d_in, const int* in_sizes, int n_in,
                              void* d_out, int out_size, void* d_ws, size_t ws_size,
                              hipStream_t stream)
{
    (void)in_sizes; (void)n_in; (void)out_size; (void)ws_size;
    const int  B = 4, S = 2048, D = 1024;
    const long BSD = (long)B * S * D;   // 8,388,608
    const long BSS = (long)B * S * S;   // 16,777,216
    const long DD  = (long)D * D;       // 1,048,576
    const int  BS  = B * S;             // 8192

    const float* x  = (const float*)d_in[0];
    // d_in[1] = causal mask (tril) -> reproduced analytically, not streamed
    const float* Wq = (const float*)d_in[2]; const float* bq = (const float*)d_in[3];
    const float* Wk = (const float*)d_in[4]; const float* bk = (const float*)d_in[5];
    const float* Wv = (const float*)d_in[6]; const float* bv = (const float*)d_in[7];
    const float* Wo = (const float*)d_in[8]; const float* bo = (const float*)d_in[9];

    float* out   = (float*)d_out;              // (B,S,D)
    float* attnF = out + BSD;                  // (B,S,S)

    bf16* xb    = (bf16*)d_ws;                 // BSD
    bf16* Wqb   = xb   + BSD;                  // DD
    bf16* Wkb   = Wqb  + DD;
    bf16* Wvb   = Wkb  + DD;
    bf16* Wob   = Wvb  + DD;
    bf16* Qb    = Wob  + DD;                   // BSD
    bf16* Kb    = Qb   + BSD;                  // BSD
    bf16* VTb   = Kb   + BSD;                  // BSD, stored transposed (D, B*S)
    bf16* attnB = VTb  + BSD;                  // BSS
    bf16* Ob    = xb;                          // attn@V result reuses xb (dead by then)

    const dim3 blk(256);

    // 0) one-shot bf16 conversion of activations + weights
    cvt4<<<(unsigned)((BSD / 4 + 255) / 256), blk, 0, stream>>>(x,  xb,  BSD / 4);
    cvt4<<<(unsigned)((DD  / 4 + 255) / 256), blk, 0, stream>>>(Wq, Wqb, DD / 4);
    cvt4<<<(unsigned)((DD  / 4 + 255) / 256), blk, 0, stream>>>(Wk, Wkb, DD / 4);
    cvt4<<<(unsigned)((DD  / 4 + 255) / 256), blk, 0, stream>>>(Wv, Wvb, DD / 4);
    cvt4<<<(unsigned)((DD  / 4 + 255) / 256), blk, 0, stream>>>(Wo, Wob, DD / 4);

    // 1) projections: (BS,D) x (D,D)^T + bias
    const dim3 g1(D / TN, BS / TM, 1);
    gemm_bf16<EPI_BIAS_BF16, false><<<g1, blk, 0, stream>>>(
        xb, Wqb, bq, nullptr, Qb, D, D, D, D, 0, 0, 0, 1.0f);
    gemm_bf16<EPI_BIAS_BF16, false><<<g1, blk, 0, stream>>>(
        xb, Wkb, bk, nullptr, Kb, D, D, D, D, 0, 0, 0, 1.0f);
    gemm_bf16<EPI_BIAS_BF16_T, false><<<g1, blk, 0, stream>>>(   // V^T: (D, B*S)
        xb, Wvb, bv, nullptr, VTb, D, D, D, BS, 0, 0, 0, 1.0f);

    // 2) scores = Q K^T / 32 with causal mask -> d_out attn region
    const dim3 g2(S / TN, S / TM, B);
    gemm_bf16<EPI_SCORES, true><<<g2, blk, 0, stream>>>(
        Qb, Kb, nullptr, attnF, nullptr, D, D, D, S,
        (long)S * D, (long)S * D, (long)S * S, 0.03125f);

    // 3) row softmax (fp32 in place + bf16 copy)
    softmax_rows<<<BS, blk, 0, stream>>>(attnF, attnB, S);

    // 4) attn @ V : B = V^T (D rows, ld = B*S, batch offset = z*S)
    const dim3 g3(D / TN, S / TM, B);
    gemm_bf16<EPI_BF16, false><<<g3, blk, 0, stream>>>(
        attnB, VTb, nullptr, nullptr, Ob, S, S, BS, D,
        (long)S * S, (long)S, (long)S * D, 1.0f);

    // 5) output projection -> fp32 d_out
    gemm_bf16<EPI_BIAS_F32, false><<<g1, blk, 0, stream>>>(
        Ob, Wob, bo, out, nullptr, D, D, D, D, 0, 0, 0, 1.0f);
}
